// CrossBlock_59871844106773
// MI455X (gfx1250) — compile-verified, matrix operand order checked by reading, not run on hardware
//
#include <hip/hip_runtime.h>
#include <hip/hip_bf16.h>
#include <math.h>

// Problem constants (reference): B=16, LQ=LKV=512, D=768, H=12, HD=64, HID=3072
#define DMODEL 768

typedef __attribute__((ext_vector_type(16))) __bf16 v16bf;
typedef __attribute__((ext_vector_type(8)))  __bf16 v8bf;
typedef __attribute__((ext_vector_type(8)))  float  v8f;

__device__ __forceinline__ unsigned short f2bf(float f) {
    __bf16 h = (__bf16)f;
    return __builtin_bit_cast(unsigned short, h);
}

// store 4 floats as 4 bf16 (8 bytes)
__device__ __forceinline__ void st_bf16x4(unsigned short* dst, float4 f) {
    unsigned lo = (unsigned)f2bf(f.x) | ((unsigned)f2bf(f.y) << 16);
    unsigned hi = (unsigned)f2bf(f.z) | ((unsigned)f2bf(f.w) << 16);
    *(uint2*)dst = make_uint2(lo, hi);
}

// LDS byte offset of a __shared__ object (LDS aperture keeps offset in low 32 bits)
__device__ __forceinline__ unsigned lds_off(const void* p) {
    return (unsigned)(uintptr_t)p;
}

// CDNA5 async copy: 16 bytes/lane global -> LDS, tracked by ASYNCcnt
__device__ __forceinline__ void async_ld16(const void* g, unsigned lds) {
    asm volatile("global_load_async_to_lds_b128 %0, %1, off"
                 :: "v"(lds), "v"(g) : "memory");
}
#define S_WAIT_ASYNC(n) asm volatile("s_wait_asynccnt " #n ::: "memory")

// Build a v16bf A/B fragment per CDNA5 16-bit WMMA layout:
// lane<16: K = koff..+7 and koff+16..+23 ; lane>=16: caller adds 8 to koff.
__device__ __forceinline__ v16bf ld_frag(const unsigned short* rowPtr, int koff) {
    union U { v16bf v; v8bf h[2]; } u;
    u.h[0] = *(const v8bf*)(rowPtr + koff);
    u.h[1] = *(const v8bf*)(rowPtr + koff + 16);
    return u.v;
}

// ---------------------------------------------------------------------------
// GEMM: out[M,N] = A[M,K](bf16) @ W[N,K](bf16)^T (+bias)(+GELU)(+res fp32)
// 256 threads (8 waves arranged 2(M) x 4(N)), block tile 128 x 128,
// wave tile 64 x 32 (acc[4][2]), K-step 32, double-buffered LDS staged via
// global_load_async_to_lds_b128 (4 async chunks/thread per slab).
// flags: 1=bias, 2=residual(fp32), 4=gelu, 8=bf16 output
// ---------------------------------------------------------------------------
__global__ __launch_bounds__(256)
void gemm_wmma_bf16(const unsigned short* __restrict__ A,
                    const unsigned short* __restrict__ W,
                    const float* __restrict__ bias, const float* __restrict__ res,
                    void* __restrict__ outv, int M, int N, int K, int flags)
{
    __shared__ unsigned short As[2][128 * 40];  // bf16 bits, row stride 40 (pad)
    __shared__ unsigned short Ws[2][128 * 40];

    const int tid  = threadIdx.x;
    const int lane = tid & 31;
    const int wave = tid >> 5;
    const int wm   = wave & 1;          // 2 waves along M (64 rows each)
    const int wn   = wave >> 1;         // 4 waves along N (32 cols each)
    const int l16  = lane & 15;
    const int halfShift = (lane >> 4) * 8;

    const int bm = blockIdx.y * 128;
    const int bn = blockIdx.x * 128;

    // async staging: A tile 128x32 = 512 16B-chunks -> 2/thread (same for W)
    const int sr = tid >> 1;            // 0..127 (row for both A and W tiles)
    const int sc = (tid & 1) * 16;      // col 0 / 16 (two 8-elem chunks each)

    const unsigned short* aRow = A + (size_t)(bm + sr) * K + sc;
    const unsigned short* wRow = W + (size_t)(bn + sr) * K + sc;

    v8f acc[4][2] = {};
    const int nk = K >> 5;

    // prologue: slab 0 -> buffer 0
    async_ld16(aRow,     lds_off(&As[0][sr * 40 + sc]));
    async_ld16(aRow + 8, lds_off(&As[0][sr * 40 + sc + 8]));
    async_ld16(wRow,     lds_off(&Ws[0][sr * 40 + sc]));
    async_ld16(wRow + 8, lds_off(&Ws[0][sr * 40 + sc + 8]));

    for (int ks = 0; ks < nk; ++ks) {
        const int buf = ks & 1;
        if (ks + 1 < nk) {
            const unsigned short* an = aRow + (size_t)(ks + 1) * 32;
            const unsigned short* wn2 = wRow + (size_t)(ks + 1) * 32;
            async_ld16(an,      lds_off(&As[buf ^ 1][sr * 40 + sc]));
            async_ld16(an + 8,  lds_off(&As[buf ^ 1][sr * 40 + sc + 8]));
            async_ld16(wn2,     lds_off(&Ws[buf ^ 1][sr * 40 + sc]));
            async_ld16(wn2 + 8, lds_off(&Ws[buf ^ 1][sr * 40 + sc + 8]));
            S_WAIT_ASYNC(4);            // slab ks landed (4 newer ops in flight)
        } else {
            S_WAIT_ASYNC(0);
        }
        __syncthreads();

        v16bf af[4], bfm[2];
        #pragma unroll
        for (int i = 0; i < 4; ++i)
            af[i] = ld_frag(&As[buf][(wm * 64 + i * 16 + l16) * 40], halfShift);
        #pragma unroll
        for (int j = 0; j < 2; ++j)
            bfm[j] = ld_frag(&Ws[buf][(wn * 32 + j * 16 + l16) * 40], halfShift);

        #pragma unroll
        for (int i = 0; i < 4; ++i)
            #pragma unroll
            for (int j = 0; j < 2; ++j)
                acc[i][j] = __builtin_amdgcn_wmma_f32_16x16x32_bf16(
                    false, af[i], false, bfm[j], (short)0, acc[i][j], false, false);
        __syncthreads();                // buffer free before reuse at ks+2
    }

    const bool doBias = flags & 1, doRes = flags & 2, doGelu = flags & 4, doBf = flags & 8;
    #pragma unroll
    for (int i = 0; i < 4; ++i) {
        #pragma unroll
        for (int j = 0; j < 2; ++j) {
            const int c = bn + wn * 32 + j * 16 + l16;
            const float bc = doBias ? bias[c] : 0.0f;
            #pragma unroll
            for (int e = 0; e < 8; ++e) {
                const int r = bm + wm * 64 + i * 16 + e + halfShift;
                float v = acc[i][j][e] + bc;
                if (doGelu) v = 0.5f * v * (1.0f + erff(v * 0.70710678118f));
                if (doRes)  v += res[(size_t)r * N + c];
                if (doBf) ((unsigned short*)outv)[(size_t)r * N + c] = f2bf(v);
                else      ((float*)outv)[(size_t)r * N + c] = v;
            }
        }
    }
}

// ---------------------------------------------------------------------------
// Flash attention (bf16 in / bf16 out) for one (b, h, 128-query tile).
// Q rows: Qg[(b*512+l)*qStride + h*64 + d]          (bf16)
// K rows: KVg[(b*512+l)*kvStride + kColOff + h*64]  (bf16)
// V rows: KVg[(b*512+l)*kvStride + vColOff + h*64]  (bf16)
// Out:    Og[(b*512+l)*768 + h*64 + d]              (bf16)
// ---------------------------------------------------------------------------
__global__ __launch_bounds__(256)
void attn_wmma_bf16(const unsigned short* __restrict__ Qg,
                    const unsigned short* __restrict__ KVg,
                    unsigned short* __restrict__ Og, int qStride, int kvStride,
                    int kColOff, int vColOff, int Lkv)
{
    __shared__ unsigned short Qs[128 * 72];
    __shared__ unsigned short Ks[64 * 72];
    __shared__ unsigned short Vt[64 * 72];   // transposed: [d][kv]
    __shared__ float          Ss[128 * 65];  // scores fp32 (bank-safe pad)
    __shared__ unsigned short Ps[128 * 72];  // probabilities bf16
    __shared__ float mrow[128], lrow[128], arow[128];

    const int b  = blockIdx.z;
    const int h  = blockIdx.y;
    const int q0 = blockIdx.x * 128;
    const int tid  = threadIdx.x;
    const int lane = tid & 31;
    const int wave = tid >> 5;
    const int wm = wave & 3, wn = wave >> 2;
    const int l16 = lane & 15;
    const int halfShift = (lane >> 4) * 8;
    const float scale = 0.125f; // 64^-0.5

    const unsigned short* Qbase = Qg  + (size_t)b * 512 * qStride + h * 64;
    const unsigned short* Kbase = KVg + (size_t)b * (size_t)Lkv * kvStride + kColOff + h * 64;
    const unsigned short* Vbase = KVg + (size_t)b * (size_t)Lkv * kvStride + vColOff + h * 64;
    unsigned short*       Obase = Og  + (size_t)b * 512 * DMODEL + h * 64;

    // stage Q tile async: 128x64 bf16 = 1024 chunks -> 4/thread
    {
        const int r = tid >> 1, c0 = (tid & 1) * 4;
        const unsigned short* src = Qbase + (size_t)(q0 + r) * qStride;
        #pragma unroll
        for (int c = 0; c < 4; ++c) {
            const int col = (c0 + c) * 8;
            async_ld16(src + col, lds_off(&Qs[r * 72 + col]));
        }
    }
    if (tid < 128) { mrow[tid] = -INFINITY; lrow[tid] = 0.0f; }

    const int rk = tid >> 2;            // 0..63 (K/V staging row)
    const int kc0 = (tid & 3) * 2;      // two 8-elem chunks
    const int vc  = (tid & 3) * 16;     // 16 cols for V transpose

    v8f o[2][2] = {};

    for (int kt = 0; kt < Lkv; kt += 64) {
        __syncthreads();                // LDS reusable (prev compute done)
        // stage K tile async: 64x64 = 512 chunks -> 2/thread
        {
            const unsigned short* src = Kbase + (size_t)(kt + rk) * kvStride;
            async_ld16(src + kc0 * 8,     lds_off(&Ks[rk * 72 + kc0 * 8]));
            async_ld16(src + kc0 * 8 + 8, lds_off(&Ks[rk * 72 + kc0 * 8 + 8]));
        }
        // stage V tile transposed (manual): Vt[d][kv]
        {
            const unsigned short* vs = Vbase + (size_t)(kt + rk) * kvStride + vc;
            uint4 w0 = *(const uint4*)vs;
            uint4 w1 = *(const uint4*)(vs + 8);
            unsigned wv[8] = {w0.x, w0.y, w0.z, w0.w, w1.x, w1.y, w1.z, w1.w};
            #pragma unroll
            for (int i = 0; i < 8; ++i) {
                Vt[(vc + 2 * i)     * 72 + rk] = (unsigned short)(wv[i] & 0xffffu);
                Vt[(vc + 2 * i + 1) * 72 + rk] = (unsigned short)(wv[i] >> 16);
            }
        }
        S_WAIT_ASYNC(0);
        __syncthreads();

        // S = scale * Q @ K^T : [128 q x 64 kv], each wave 32x32
        v8f s[2][2] = {};
        #pragma unroll
        for (int d0 = 0; d0 < 64; d0 += 32) {
            const int ko = d0 + halfShift;
            v16bf qf[2], kf[2];
            #pragma unroll
            for (int i = 0; i < 2; ++i)
                qf[i] = ld_frag(&Qs[(wm * 32 + i * 16 + l16) * 72], ko);
            #pragma unroll
            for (int j = 0; j < 2; ++j)
                kf[j] = ld_frag(&Ks[(wn * 32 + j * 16 + l16) * 72], ko);
            #pragma unroll
            for (int i = 0; i < 2; ++i)
                #pragma unroll
                for (int j = 0; j < 2; ++j)
                    s[i][j] = __builtin_amdgcn_wmma_f32_16x16x32_bf16(
                        false, qf[i], false, kf[j], (short)0, s[i][j], false, false);
        }
        #pragma unroll
        for (int i = 0; i < 2; ++i)
            #pragma unroll
            for (int j = 0; j < 2; ++j)
                #pragma unroll
                for (int e = 0; e < 8; ++e)
                    Ss[(wm * 32 + i * 16 + e + halfShift) * 65 + wn * 32 + j * 16 + l16]
                        = s[i][j][e] * scale;
        __syncthreads();

        // online softmax: one thread per query row
        if (tid < 128) {
            const float* sr = &Ss[tid * 65];
            float mOld = mrow[tid], mNew = mOld;
            #pragma unroll 8
            for (int k = 0; k < 64; ++k) mNew = fmaxf(mNew, sr[k]);
            const float alpha = expf(mOld - mNew);
            float sum = 0.0f;
            unsigned short* pr = &Ps[tid * 72];
            #pragma unroll 8
            for (int k = 0; k < 64; ++k) {
                float p = expf(sr[k] - mNew);
                pr[k] = f2bf(p);
                sum += p;
            }
            lrow[tid] = lrow[tid] * alpha + sum;
            mrow[tid] = mNew;
            arow[tid] = alpha;
        }
        __syncthreads();

        // O = O*alpha + P @ V
        #pragma unroll
        for (int i = 0; i < 2; ++i) {
            float al[8];
            #pragma unroll
            for (int e = 0; e < 8; ++e)
                al[e] = arow[wm * 32 + i * 16 + e + halfShift];
            #pragma unroll
            for (int j = 0; j < 2; ++j)
                #pragma unroll
                for (int e = 0; e < 8; ++e)
                    o[i][j][e] *= al[e];
        }
        #pragma unroll
        for (int ksl = 0; ksl < 64; ksl += 32) {
            const int ko = ksl + halfShift;
            v16bf pf[2], vf[2];
            #pragma unroll
            for (int i = 0; i < 2; ++i)
                pf[i] = ld_frag(&Ps[(wm * 32 + i * 16 + l16) * 72], ko);
            #pragma unroll
            for (int j = 0; j < 2; ++j)
                vf[j] = ld_frag(&Vt[(wn * 32 + j * 16 + l16) * 72], ko);
            #pragma unroll
            for (int i = 0; i < 2; ++i)
                #pragma unroll
                for (int j = 0; j < 2; ++j)
                    o[i][j] = __builtin_amdgcn_wmma_f32_16x16x32_bf16(
                        false, pf[i], false, vf[j], (short)0, o[i][j], false, false);
        }
    }

    // normalize by 1/l and write bf16
    #pragma unroll
    for (int i = 0; i < 2; ++i) {
        #pragma unroll
        for (int j = 0; j < 2; ++j) {
            const int dcol = wn * 32 + j * 16 + l16;
            #pragma unroll
            for (int e = 0; e < 8; ++e) {
                const int rloc = wm * 32 + i * 16 + e + halfShift;
                const float inv = 1.0f / lrow[rloc];
                Obase[(size_t)(q0 + rloc) * DMODEL + dcol] = f2bf(o[i][j][e] * inv);
            }
        }
    }
}

// ---------------------------------------------------------------------------
// LayerNorm (fp32 in -> bf16 out): one block per row
// ---------------------------------------------------------------------------
__global__ __launch_bounds__(256)
void ln_kernel(const float* __restrict__ x, const float* __restrict__ g,
               const float* __restrict__ bta, unsigned short* __restrict__ y, int D)
{
    __shared__ float red[256];
    const size_t row = blockIdx.x;
    const float* xr = x + row * D;
    const int tid = threadIdx.x;

    float s = 0.0f, s2 = 0.0f;
    for (int i = tid; i < D; i += 256) { float v = xr[i]; s += v; s2 += v * v; }

    red[tid] = s; __syncthreads();
    for (int o = 128; o > 0; o >>= 1) { if (tid < o) red[tid] += red[tid + o]; __syncthreads(); }
    const float mean = red[0] / (float)D;
    __syncthreads();
    red[tid] = s2; __syncthreads();
    for (int o = 128; o > 0; o >>= 1) { if (tid < o) red[tid] += red[tid + o]; __syncthreads(); }
    const float var = red[0] / (float)D - mean * mean;
    const float inv = rsqrtf(var + 1e-5f);

    unsigned short* yr = y + row * D;
    for (int i = tid; i < D; i += 256)
        yr[i] = f2bf((xr[i] - mean) * inv * g[i] + bta[i]);
}

// fp32 -> bf16 bulk convert (weights), n % 4 == 0
__global__ __launch_bounds__(256)
void cvt_bf16(const float* __restrict__ x, unsigned short* __restrict__ y, int n)
{
    const int i = (blockIdx.x * 256 + threadIdx.x) * 4;
    if (i < n) st_bf16x4(y + i, *(const float4*)(x + i));
}

// ---------------------------------------------------------------------------
extern "C" void kernel_launch(void* const* d_in, const int* in_sizes, int n_in,
                              void* d_out, int out_size, void* d_ws, size_t ws_size,
                              hipStream_t stream) {
    (void)in_sizes; (void)n_in; (void)out_size; (void)ws_size;
    const float* q     = (const float*)d_in[0];
    const float* kv    = (const float*)d_in[1];
    const float* n1g   = (const float*)d_in[2];
    const float* n1b   = (const float*)d_in[3];
    const float* qkv_w = (const float*)d_in[4];
    const float* sa_w  = (const float*)d_in[5];
    const float* sa_b  = (const float*)d_in[6];
    const float* n2qg  = (const float*)d_in[7];
    const float* n2qb  = (const float*)d_in[8];
    const float* n2kg  = (const float*)d_in[9];
    const float* n2kb  = (const float*)d_in[10];
    const float* caq_w = (const float*)d_in[11];
    const float* cakv_w= (const float*)d_in[12];
    const float* cap_w = (const float*)d_in[13];
    const float* cap_b = (const float*)d_in[14];
    const float* n3g   = (const float*)d_in[15];
    const float* n3b   = (const float*)d_in[16];
    const float* f1w   = (const float*)d_in[17];
    const float* f1b   = (const float*)d_in[18];
    const float* f2w   = (const float*)d_in[19];
    const float* f2b   = (const float*)d_in[20];
    float* out = (float*)d_out;

    const size_t SZ = (size_t)8192 * 768;  // B*L x D
    unsigned short* u = (unsigned short*)d_ws;
    unsigned short* A_bf  = u;                     // ln outs / attn outs
    unsigned short* F_bf  = A_bf + SZ;             // xkv
    unsigned short* E_bf  = F_bf + SZ;             // qh
    unsigned short* Bq_bf = E_bf + SZ;             // qkv / kvp / mlp hidden (8192*3072)
    unsigned short* Wq   = Bq_bf + (size_t)8192 * 3072;
    unsigned short* Wsa  = Wq   + (size_t)2304 * 768;
    unsigned short* Wcq  = Wsa  + (size_t)768 * 768;
    unsigned short* Wckv = Wcq  + (size_t)768 * 768;
    unsigned short* Wcp  = Wckv + (size_t)1536 * 768;
    unsigned short* Wf1  = Wcp  + (size_t)768 * 768;
    unsigned short* Wf2  = Wf1  + (size_t)3072 * 768;
    float* C_f = (float*)(Wf2 + (size_t)768 * 3072);   // q1
    float* D_f = C_f + SZ;                             // q2

    const dim3 blk(256);
    const dim3 attnGrid(4, 12, 16);

    // weight conversion (bf16)
    auto cvt = [&](const float* src, unsigned short* dst, int n) {
        cvt_bf16<<<(n / 4 + 255) / 256, blk, 0, stream>>>(src, dst, n);
    };
    cvt(qkv_w, Wq,   2304 * 768);
    cvt(sa_w,  Wsa,  768 * 768);
    cvt(caq_w, Wcq,  768 * 768);
    cvt(cakv_w,Wckv, 1536 * 768);
    cvt(cap_w, Wcp,  768 * 768);
    cvt(f1w,   Wf1,  3072 * 768);
    cvt(f2w,   Wf2,  768 * 3072);

    // --- self-attention ---
    ln_kernel<<<8192, blk, 0, stream>>>(q, n1g, n1b, A_bf, 768);
    gemm_wmma_bf16<<<dim3(18, 64), blk, 0, stream>>>(A_bf, Wq, nullptr, nullptr, Bq_bf, 8192, 2304, 768, 8);
    attn_wmma_bf16<<<attnGrid, blk, 0, stream>>>(Bq_bf, Bq_bf, A_bf, 2304, 2304, 768, 1536, 512);
    gemm_wmma_bf16<<<dim3(6, 64), blk, 0, stream>>>(A_bf, Wsa, sa_b, q, C_f, 8192, 768, 768, 1 | 2);
    // --- cross-attention ---
    ln_kernel<<<8192, blk, 0, stream>>>(C_f, n2qg, n2qb, A_bf, 768);
    ln_kernel<<<8192, blk, 0, stream>>>(kv, n2kg, n2kb, F_bf, 768);
    gemm_wmma_bf16<<<dim3(6, 64), blk, 0, stream>>>(A_bf, Wcq, nullptr, nullptr, E_bf, 8192, 768, 768, 8);
    gemm_wmma_bf16<<<dim3(12, 64), blk, 0, stream>>>(F_bf, Wckv, nullptr, nullptr, Bq_bf, 8192, 1536, 768, 8);
    attn_wmma_bf16<<<attnGrid, blk, 0, stream>>>(E_bf, Bq_bf, A_bf, 768, 1536, 0, 768, 512);
    gemm_wmma_bf16<<<dim3(6, 64), blk, 0, stream>>>(A_bf, Wcp, cap_b, C_f, D_f, 8192, 768, 768, 1 | 2);
    // --- MLP ---
    ln_kernel<<<8192, blk, 0, stream>>>(D_f, n3g, n3b, A_bf, 768);
    gemm_wmma_bf16<<<dim3(24, 64), blk, 0, stream>>>(A_bf, Wf1, f1b, nullptr, Bq_bf, 8192, 3072, 768, 1 | 4 | 8);
    gemm_wmma_bf16<<<dim3(6, 64), blk, 0, stream>>>(Bq_bf, Wf2, f2b, D_f, out, 8192, 768, 3072, 1 | 2);
}
